// GraphAttentionLayer_81527069213079
// MI455X (gfx1250) — compile-verified, matrix operand order checked by reading, not run on hardware
//
#include <hip/hip_runtime.h>
#include <hip/hip_bf16.h>

#define N_NODES 100000
#define N_EDGES 1600000
#define IN_F 256
#define OUT_F 64
#define ALPHA 0.2f

typedef __attribute__((ext_vector_type(16))) __bf16 v16bf;
typedef __attribute__((ext_vector_type(8)))  float  v8f;

// Order-preserving float <-> uint encoding so atomicMax(u32) == float max.
__device__ __forceinline__ unsigned ordenc(float f) {
  unsigned u = __float_as_uint(f);
  return (u & 0x80000000u) ? ~u : (u | 0x80000000u);
}
__device__ __forceinline__ float orddec(unsigned e) {
  unsigned b = (e & 0x80000000u) ? (e & 0x7FFFFFFFu) : ~e;
  return __uint_as_float(b);
}

__global__ void k_init(float* __restrict__ out, unsigned* __restrict__ m,
                       float* __restrict__ denom) {
  size_t tid = (size_t)blockIdx.x * blockDim.x + threadIdx.x;
  if (tid < (size_t)N_NODES * OUT_F) out[tid] = 0.0f;
  if (tid < N_NODES) { m[tid] = 0u; denom[tid] = 0.0f; }  // 0u == ordenc identity for max
}

// h = x @ W via v_wmma_f32_16x16x32_bf16, fused s1 = h@a1, s2 = h@a2.
// Wave computes a 16x64 tile: 4 col-groups x 8 K-steps = 32 WMMAs.
__global__ void __launch_bounds__(256) k_gemm(
    const float* __restrict__ x, const float* __restrict__ W,
    const float* __restrict__ a, float* __restrict__ h,
    float* __restrict__ s1, float* __restrict__ s2) {
  // W staged once per block into LDS, pre-swizzled to exact B-fragment layout:
  // fragment f = kiter*4+cg, 512 bf16 each; lane's 16 values contiguous (32B).
  __shared__ alignas(32) __bf16 Wswz[IN_F * OUT_F];  // 32 KB of the 320 KB WGP LDS
  const int t = threadIdx.x;
#pragma unroll
  for (int i = 0; i < 64; ++i) {
    int p = t + i * 256;
    int f = p >> 9, r = p & 511;
    int ln = r >> 4, j = r & 15;
    int kiter = f >> 2, cg = f & 3;
    int kh = ln >> 4, l16 = ln & 15;
    int K = ((j < 8) ? j : j + 8) + kh * 8;   // ISA 16-bit operand K mapping
    Wswz[p] = (__bf16)W[(kiter * 32 + K) * OUT_F + cg * 16 + l16];
  }
  __syncthreads();

  const int wave  = t >> 5;
  const int lane  = t & 31;
  const int l16   = lane & 15;
  const int khalf = lane >> 4;
  const int rowBase = blockIdx.x * 128 + wave * 16;
  int rowA = rowBase + l16;
  if (rowA >= N_NODES) rowA = N_NODES - 1;          // clamp loads, mask stores
  const float* xr = x + (size_t)rowA * IN_F + khalf * 8;

  v8f acc0 = {}, acc1 = {}, acc2 = {}, acc3 = {};
#pragma unroll
  for (int kiter = 0; kiter < 8; ++kiter) {
    // A fragment: two contiguous 8-float runs per lane (coalesced float4 x4)
    const float4 xa = *(const float4*)(xr + kiter * 32 + 0);
    const float4 xb = *(const float4*)(xr + kiter * 32 + 4);
    const float4 xc = *(const float4*)(xr + kiter * 32 + 16);
    const float4 xd = *(const float4*)(xr + kiter * 32 + 20);
    v16bf af;
    af[0]=(__bf16)xa.x;  af[1]=(__bf16)xa.y;  af[2]=(__bf16)xa.z;  af[3]=(__bf16)xa.w;
    af[4]=(__bf16)xb.x;  af[5]=(__bf16)xb.y;  af[6]=(__bf16)xb.z;  af[7]=(__bf16)xb.w;
    af[8]=(__bf16)xc.x;  af[9]=(__bf16)xc.y;  af[10]=(__bf16)xc.z; af[11]=(__bf16)xc.w;
    af[12]=(__bf16)xd.x; af[13]=(__bf16)xd.y; af[14]=(__bf16)xd.z; af[15]=(__bf16)xd.w;

    const v16bf* wp = (const v16bf*)(Wswz + (size_t)(kiter * 4) * 512);
    v16bf b0 = wp[0 * 32 + lane];
    v16bf b1 = wp[1 * 32 + lane];
    v16bf b2 = wp[2 * 32 + lane];
    v16bf b3 = wp[3 * 32 + lane];
    acc0 = __builtin_amdgcn_wmma_f32_16x16x32_bf16(false, af, false, b0, (short)0, acc0, false, false);
    acc1 = __builtin_amdgcn_wmma_f32_16x16x32_bf16(false, af, false, b1, (short)0, acc1, false, false);
    acc2 = __builtin_amdgcn_wmma_f32_16x16x32_bf16(false, af, false, b2, (short)0, acc2, false, false);
    acc3 = __builtin_amdgcn_wmma_f32_16x16x32_bf16(false, af, false, b3, (short)0, acc3, false, false);
  }

  // epilogue: store h tile + fused s1/s2 (wave-shuffle reduction over 16 cols/half)
  float a1v[4], a2v[4];
#pragma unroll
  for (int cg = 0; cg < 4; ++cg) {
    a1v[cg] = a[cg * 16 + l16];
    a2v[cg] = a[OUT_F + cg * 16 + l16];
  }
#pragma unroll
  for (int r = 0; r < 8; ++r) {
    int row = rowBase + r + khalf * 8;   // C/D layout: lanes16-31 hold M=r+8
    bool ok = row < N_NODES;
    float c0 = acc0[r], c1 = acc1[r], c2 = acc2[r], c3 = acc3[r];
    if (ok) {
      float* hp = h + (size_t)row * OUT_F + l16;
      hp[0] = c0; hp[16] = c1; hp[32] = c2; hp[48] = c3;
    }
    float p1 = c0 * a1v[0] + c1 * a1v[1] + c2 * a1v[2] + c3 * a1v[3];
    float p2 = c0 * a2v[0] + c1 * a2v[1] + c2 * a2v[2] + c3 * a2v[3];
#pragma unroll
    for (int mk = 1; mk < 16; mk <<= 1) {   // stays within each 16-lane half
      p1 += __shfl_xor(p1, mk, 32);
      p2 += __shfl_xor(p2, mk, 32);
    }
    if (ok && l16 == 0) { s1[row] = p1; s2[row] = p2; }
  }
}

__global__ void k_edge_max(const int* __restrict__ src, const int* __restrict__ dst,
                           const float* __restrict__ s1, const float* __restrict__ s2,
                           unsigned* __restrict__ m) {
  int e = blockIdx.x * blockDim.x + threadIdx.x;
  if (e >= N_EDGES) return;
  int s = src[e], d = dst[e];
  float l = s1[s] + s2[d];
  float lr = l > 0.f ? l : ALPHA * l;
  atomicMax(&m[s], ordenc(lr));
}

__global__ void k_edge_sum(const int* __restrict__ src, const int* __restrict__ dst,
                           const float* __restrict__ s1, const float* __restrict__ s2,
                           const unsigned* __restrict__ m, float* __restrict__ denom) {
  int e = blockIdx.x * blockDim.x + threadIdx.x;
  if (e >= N_EDGES) return;
  int s = src[e], d = dst[e];
  float l = s1[s] + s2[d];
  float lr = l > 0.f ? l : ALPHA * l;
  unsafeAtomicAdd(&denom[s], __expf(lr - orddec(m[s])));
}

// One wave per edge: lane covers 2 features -> 256B coalesced gather + atomics (L2-resident).
__global__ void __launch_bounds__(256) k_edge_aggr(
    const int* __restrict__ src, const int* __restrict__ dst,
    const float* __restrict__ s1, const float* __restrict__ s2,
    const unsigned* __restrict__ m, const float* __restrict__ denom,
    const float* __restrict__ h, float* __restrict__ out) {
  int wave = threadIdx.x >> 5;
  int lane = threadIdx.x & 31;
  long e = (long)blockIdx.x * 8 + wave;
  if (e >= N_EDGES) return;
  int s = src[e], d = dst[e];
  float l = s1[s] + s2[d];
  float lr = l > 0.f ? l : ALPHA * l;
  float att = __expf(lr - orddec(m[s])) / denom[s];
  const float2 hv = *(const float2*)(h + (size_t)d * OUT_F + lane * 2);
  float* op = out + (size_t)s * OUT_F + lane * 2;
  unsafeAtomicAdd(op,     att * hv.x);
  unsafeAtomicAdd(op + 1, att * hv.y);
}

__global__ void k_elu(float* __restrict__ out) {
  size_t i = (size_t)blockIdx.x * blockDim.x + threadIdx.x;
  if (i < (size_t)N_NODES * OUT_F) {
    float v = out[i];
    out[i] = v > 0.f ? v : __expf(v) - 1.f;
  }
}

extern "C" void kernel_launch(void* const* d_in, const int* in_sizes, int n_in,
                              void* d_out, int out_size, void* d_ws, size_t ws_size,
                              hipStream_t stream) {
  const float* x       = (const float*)d_in[0];
  const int*   indices = (const int*)d_in[1];
  const float* W       = (const float*)d_in[2];
  const float* a       = (const float*)d_in[3];
  float* out = (float*)d_out;

  float*    h     = (float*)d_ws;                       // N*64 f32 = 25.6 MB
  float*    s1    = h + (size_t)N_NODES * OUT_F;        // N f32
  float*    s2    = s1 + N_NODES;                       // N f32
  unsigned* m     = (unsigned*)(s2 + N_NODES);          // N u32
  float*    denom = (float*)(m + N_NODES);              // N f32

  const int* src = indices;
  const int* dst = indices + N_EDGES;

  const int feat_total = N_NODES * OUT_F;
  k_init<<<(feat_total + 255) / 256, 256, 0, stream>>>(out, m, denom);
  k_gemm<<<(N_NODES + 127) / 128, 256, 0, stream>>>(x, W, a, h, s1, s2);
  k_edge_max<<<(N_EDGES + 255) / 256, 256, 0, stream>>>(src, dst, s1, s2, m);
  k_edge_sum<<<(N_EDGES + 255) / 256, 256, 0, stream>>>(src, dst, s1, s2, m, denom);
  k_edge_aggr<<<(N_EDGES + 7) / 8, 256, 0, stream>>>(src, dst, s1, s2, m, denom, h, out);
  k_elu<<<(feat_total + 255) / 256, 256, 0, stream>>>(out);
}
